// GATE_34935263986009
// MI455X (gfx1250) — compile-verified, matrix operand order checked by reading, not run on hardware
//
#include <hip/hip_runtime.h>
#include <math.h>

typedef __attribute__((ext_vector_type(16))) _Float16 v16h;
typedef __attribute__((ext_vector_type(8)))  _Float16 v8h;
typedef __attribute__((ext_vector_type(8)))  float    v8f;

constexpr int N_NODES = 50000;
constexpr int N_EDGES = 1600000;
constexpr int EP      = N_EDGES + N_NODES;   // edges + self loops
constexpr int F_IN    = 512;
constexpr int F_HID   = 128;                 // HIDIM == OUTDIM == 128
constexpr int MT_ROWS = 80;                  // 5 x 16-row tiles per block; 50000 = 80*625
constexpr float NEG_SLOPE = 0.2f;

// ---------------- helpers ----------------

// order-preserving float <-> uint encoding (for exact atomic segment-max)
__device__ __forceinline__ unsigned fenc(float x) {
    unsigned b = __float_as_uint(x);
    return b ^ ((b >> 31) ? 0xFFFFFFFFu : 0x80000000u);
}
__device__ __forceinline__ float fdec(unsigned k) {
    unsigned b = k ^ ((k >> 31) ? 0x80000000u : 0xFFFFFFFFu);
    return __uint_as_float(b);
}

// glue two contiguous 8-half runs into one 16-half WMMA fragment.
// 16-bit fragment K-map (wave32): lane-group g holds K = 8g..8g+7 in VGPR0-3
// and K = 16+8g..16+8g+7 in VGPR4-7, sequential within each run.
__device__ __forceinline__ v16h frag16(v8h lo, v8h hi) {
    return __builtin_shufflevector(lo, hi, 0, 1, 2, 3, 4, 5, 6, 7,
                                           8, 9, 10, 11, 12, 13, 14, 15);
}

// ---------------- W pre-transpose + f16 convert: Wt[n][k] = (f16) W[k][n] ----------------
__global__ void transpose_half(const float* __restrict__ W, _Float16* __restrict__ Wt, int K) {
    int o = blockIdx.x * blockDim.x + threadIdx.x;
    if (o >= K * 128) return;
    int n = o / K, k = o - n * K;
    Wt[o] = (_Float16)W[(size_t)k * 128 + n];
}

// ---------------- WMMA GEMM: C[M x 128] = A[M x K] * Bt^T, Bt = f16 [128 x K] ----------------
// block = 256 threads = 8 waves. Block owns an 80-row stripe (5 M-tiles), wave w owns
// N-tile w. A staged in LDS as f16 (converted once, shared by all 8 waves); one B
// fragment per wave per k-step feeds 5 v_wmma ops. No divergence -> EXEC all ones.
__global__ void __launch_bounds__(256)
wmma_gemm128(const float* __restrict__ A, const _Float16* __restrict__ Bt,
             float* __restrict__ C, int K) {
    __shared__ _Float16 Ah[MT_ROWS * 32];     // 80 x 32 halves = 5 KB
    const int tid  = threadIdx.x;
    const int lane = tid & 31;
    const int wid  = tid >> 5;
    const int g    = lane >> 4;
    const int mloc = lane & 15;
    const int n    = (wid << 4) + mloc;       // output column (0..127)
    const int m0   = blockIdx.x * MT_ROWS;    // output row base

    const _Float16* brow = Bt + (size_t)n * K;

    v8f acc[5] = {};
    for (int k0 = 0; k0 < K; k0 += 32) {
        // cooperative load+convert of the 80x32 A tile: 1280 float2, 5 per thread
#pragma unroll
        for (int i = 0; i < 5; ++i) {
            int j = tid + 256 * i;            // float2 index
            int r = j >> 4;                   // 16 float2 per row
            int c = (j & 15) << 1;
            float2 v = *(const float2*)(A + (size_t)(m0 + r) * K + k0 + c);
            Ah[r * 32 + c]     = (_Float16)v.x;
            Ah[r * 32 + c + 1] = (_Float16)v.y;
        }
        __syncthreads();

        // B fragment: two contiguous 16B runs from the transposed f16 weights
        v8h blo = *(const v8h*)(brow + k0 + 8 * g);
        v8h bhi = *(const v8h*)(brow + k0 + 16 + 8 * g);
        v16h bf = frag16(blo, bhi);

#pragma unroll
        for (int mt = 0; mt < 5; ++mt) {
            int m = mt * 16 + mloc;
            v8h alo = *(const v8h*)(Ah + m * 32 + 8 * g);
            v8h ahi = *(const v8h*)(Ah + m * 32 + 16 + 8 * g);
            v16h af = frag16(alo, ahi);
            acc[mt] = __builtin_amdgcn_wmma_f32_16x16x32_f16(
                          false, af, false, bf, (short)0, acc[mt], false, false);
        }
        __syncthreads();
    }
#pragma unroll
    for (int mt = 0; mt < 5; ++mt) {
#pragma unroll
        for (int r = 0; r < 8; ++r) {
            int row = m0 + mt * 16 + r + (g << 3);   // C/D layout: M = r + 8*g
            C[(size_t)row * 128 + n] = acc[mt][r];
        }
    }
}

// ---------------- per-node attention scalars: aS[i]=h[i]·a_src, aD[i]=h[i]·a_dst ----------------
__global__ void node_alpha(const float* __restrict__ h,
                           const float* __restrict__ asrc,
                           const float* __restrict__ adst,
                           float* __restrict__ aS, float* __restrict__ aD) {
    int gt   = blockIdx.x * blockDim.x + threadIdx.x;
    int node = gt >> 5, lane = gt & 31;
    if (node >= N_NODES) return;
    float4 hv = ((const float4*)h)[node * 32 + lane];
    float4 av = ((const float4*)asrc)[lane];
    float4 dv = ((const float4*)adst)[lane];
    float s = hv.x * av.x + hv.y * av.y + hv.z * av.z + hv.w * av.w;
    float d = hv.x * dv.x + hv.y * dv.y + hv.z * dv.z + hv.w * dv.w;
    for (int off = 16; off > 0; off >>= 1) {
        s += __shfl_xor(s, off, 32);
        d += __shfl_xor(d, off, 32);
    }
    if (lane == 0) { aS[node] = s; aD[node] = d; }
}

// ---------------- deterministic reductions for mean(edge_weights) and c_e = W_e·a_e ----------------
__global__ void reduce_partial(const float* __restrict__ ew, float* __restrict__ partials) {
    __shared__ float sm[256];
    int t = threadIdx.x;
    float s = 0.f;
    for (int i = blockIdx.x * 256 + t; i < N_EDGES; i += 1024 * 256) s += ew[i];
    sm[t] = s; __syncthreads();
    for (int off = 128; off > 0; off >>= 1) { if (t < off) sm[t] += sm[t + off]; __syncthreads(); }
    if (t == 0) partials[blockIdx.x] = sm[0];
}

__global__ void finalize_scalars(const float* __restrict__ partials,
                                 const float* __restrict__ We1, const float* __restrict__ ae1,
                                 const float* __restrict__ We2, const float* __restrict__ ae2,
                                 float* __restrict__ scal) {
    __shared__ float sm[256];
    int t = threadIdx.x;
    float s = 0.f;
    for (int i = t; i < 1024; i += 256) s += partials[i];
    sm[t] = s; __syncthreads();
    for (int off = 128; off > 0; off >>= 1) { if (t < off) sm[t] += sm[t + off]; __syncthreads(); }
    if (t == 0) scal[0] = sm[0] / (float)N_EDGES;          // mean edge weight (self-loop fill)
    __syncthreads();
    sm[t] = (t < 128) ? We1[t] * ae1[t] : 0.f; __syncthreads();
    for (int off = 128; off > 0; off >>= 1) { if (t < off) sm[t] += sm[t + off]; __syncthreads(); }
    if (t == 0) scal[1] = sm[0];                           // c_e layer 1
    __syncthreads();
    sm[t] = (t < 128) ? We2[t] * ae2[t] : 0.f; __syncthreads();
    for (int off = 128; off > 0; off >>= 1) { if (t < off) sm[t] += sm[t + off]; __syncthreads(); }
    if (t == 0) scal[2] = sm[0];                           // c_e layer 2
}

// ---------------- edge passes ----------------
__global__ void edge_logits(const long long* __restrict__ ei, const float* __restrict__ ew,
                            const float* __restrict__ scal, int ce_idx,
                            const float* __restrict__ aS, const float* __restrict__ aD,
                            float* __restrict__ logits, unsigned* __restrict__ mkey) {
    int e = blockIdx.x * blockDim.x + threadIdx.x;
    if (e >= EP) return;
    int src, dst; float ea;
    if (e < N_EDGES) { src = (int)ei[e]; dst = (int)ei[N_EDGES + e]; ea = ew[e]; }
    else             { src = dst = e - N_EDGES;                      ea = scal[0]; }
    float t = aS[src] + aD[dst] + scal[ce_idx] * ea;
    t = t > 0.f ? t : NEG_SLOPE * t;                       // leaky_relu
    logits[e] = t;
    atomicMax(&mkey[dst], fenc(t));                        // exact segment-max
}

__global__ void edge_exp(const long long* __restrict__ ei,
                         float* __restrict__ logits, const unsigned* __restrict__ mkey,
                         float* __restrict__ denom) {
    int e = blockIdx.x * blockDim.x + threadIdx.x;
    if (e >= EP) return;
    int dst = (e < N_EDGES) ? (int)ei[N_EDGES + e] : (e - N_EDGES);
    float p = __expf(logits[e] - fdec(mkey[dst]));
    logits[e] = p;                                         // reuse buffer for p
    atomicAdd(&denom[dst], p);
}

// one edge handled by 32 lanes; each lane owns 4 contiguous channels (coalesced float4 row read)
__global__ void edge_scatter(const long long* __restrict__ ei,
                             const float* __restrict__ p, const float* __restrict__ denom,
                             const float* __restrict__ h, float* __restrict__ agg) {
    int t = blockIdx.x * blockDim.x + threadIdx.x;
    if (t >= EP * 32) return;
    int e = t >> 5, q = t & 31;
    int src, dst;
    if (e < N_EDGES) { src = (int)ei[e]; dst = (int)ei[N_EDGES + e]; }
    else             { src = dst = e - N_EDGES; }
    float coef = p[e] / (denom[dst] + 1e-16f);
    float4 hv = ((const float4*)h)[src * 32 + q];
    float* o = agg + (size_t)dst * 128 + q * 4;
    atomicAdd(o + 0, coef * hv.x);
    atomicAdd(o + 1, coef * hv.y);
    atomicAdd(o + 2, coef * hv.z);
    atomicAdd(o + 3, coef * hv.w);
}

// ---------------- elementwise tails ----------------
__global__ void relu_bias(const float* __restrict__ agg, const float* __restrict__ b,
                          float* __restrict__ out) {
    int i = blockIdx.x * blockDim.x + threadIdx.x;
    if (i >= N_NODES * 128) return;
    float v = agg[i] + b[i & 127];
    out[i] = v > 0.f ? v : 0.f;
}

__global__ void bias_add(float* __restrict__ out, const float* __restrict__ b) {
    int i = blockIdx.x * blockDim.x + threadIdx.x;
    if (i >= N_NODES * 128) return;
    out[i] += b[i & 127];
}

// ---------------- launch ----------------
extern "C" void kernel_launch(void* const* d_in, const int* in_sizes, int n_in,
                              void* d_out, int out_size, void* d_ws, size_t ws_size,
                              hipStream_t stream) {
    const float*     x   = (const float*)d_in[0];
    const long long* ei  = (const long long*)d_in[1];   // int64 [2, E]
    const float*     ew  = (const float*)d_in[2];
    const float*     W1  = (const float*)d_in[3];
    const float*     as1 = (const float*)d_in[4];
    const float*     ad1 = (const float*)d_in[5];
    const float*     We1 = (const float*)d_in[6];
    const float*     ae1 = (const float*)d_in[7];
    const float*     b1  = (const float*)d_in[8];
    const float*     W2  = (const float*)d_in[9];
    const float*     as2 = (const float*)d_in[10];
    const float*     ad2 = (const float*)d_in[11];
    const float*     We2 = (const float*)d_in[12];
    const float*     ae2 = (const float*)d_in[13];
    const float*     b2  = (const float*)d_in[14];
    float* out = (float*)d_out;
    char*  ws  = (char*)d_ws;

    // workspace layout (bytes)
    const size_t HBYTES = (size_t)N_NODES * 128 * 4;   // 25.6 MB
    const size_t EBYTES = (size_t)EP * 4;              // 6.6 MB
    const size_t NBYTES = (size_t)N_NODES * 4;         // 200 KB
    float*    scal     = (float*)(ws + 0);             // [mean, ce1, ce2]
    float*    partials = (float*)(ws + 256);           // 1024 floats
    float*    hbuf     = (float*)(ws + 8192);                      // h1, then relu(agg+b1)
    float*    agg      = (float*)(ws + 8192 + HBYTES);             // layer1 agg, then h2
    float*    logits   = (float*)(ws + 8192 + 2 * HBYTES);         // logits -> p
    float*    aS       = (float*)(ws + 8192 + 2 * HBYTES + EBYTES);
    float*    aD       = (float*)(ws + 8192 + 2 * HBYTES + EBYTES + NBYTES);
    unsigned* mkey     = (unsigned*)(ws + 8192 + 2 * HBYTES + EBYTES + 2 * NBYTES);
    float*    denom    = (float*)(ws + 8192 + 2 * HBYTES + EBYTES + 3 * NBYTES);
    _Float16* Wt1      = (_Float16*)(ws + 8192 + 2 * HBYTES + EBYTES + 4 * NBYTES);
    _Float16* Wt2      = (_Float16*)(ws + 8192 + 2 * HBYTES + EBYTES + 4 * NBYTES
                                        + (size_t)128 * F_IN * 2);

    const int BE  = (EP + 255) / 256;           // edge-wise blocks
    const int BS  = (EP * 32 + 255) / 256;      // scatter blocks
    const int BN  = (N_NODES * 128 + 255) / 256;
    const int BA  = (N_NODES * 32 + 255) / 256; // node_alpha blocks
    const int BG  = N_NODES / MT_ROWS;          // 625 GEMM blocks (exact)

    // zero state for this call (graph-capture safe)
    hipMemsetAsync(agg,   0, HBYTES, stream);
    hipMemsetAsync(mkey,  0, NBYTES, stream);
    hipMemsetAsync(denom, 0, NBYTES, stream);
    hipMemsetAsync(out,   0, (size_t)out_size * 4, stream);

    // scalars + weight prep
    reduce_partial<<<1024, 256, 0, stream>>>(ew, partials);
    finalize_scalars<<<1, 256, 0, stream>>>(partials, We1, ae1, We2, ae2, scal);
    transpose_half<<<(128 * F_IN + 255) / 256, 256, 0, stream>>>(W1, Wt1, F_IN);
    transpose_half<<<(128 * F_HID + 255) / 256, 256, 0, stream>>>(W2, Wt2, F_HID);

    // ---- layer 1 ----
    wmma_gemm128<<<BG, 256, 0, stream>>>(x, Wt1, hbuf, F_IN);
    node_alpha<<<BA, 256, 0, stream>>>(hbuf, as1, ad1, aS, aD);
    edge_logits<<<BE, 256, 0, stream>>>(ei, ew, scal, 1, aS, aD, logits, mkey);
    edge_exp<<<BE, 256, 0, stream>>>(ei, logits, mkey, denom);
    edge_scatter<<<BS, 256, 0, stream>>>(ei, logits, denom, hbuf, agg);
    relu_bias<<<BN, 256, 0, stream>>>(agg, b1, hbuf);   // hbuf = relu(agg + b1)

    // ---- layer 2 ----
    hipMemsetAsync(mkey,  0, NBYTES, stream);
    hipMemsetAsync(denom, 0, NBYTES, stream);
    wmma_gemm128<<<BG, 256, 0, stream>>>(hbuf, Wt2, agg, F_HID);   // agg = h2
    node_alpha<<<BA, 256, 0, stream>>>(agg, as2, ad2, aS, aD);
    edge_logits<<<BE, 256, 0, stream>>>(ei, ew, scal, 2, aS, aD, logits, mkey);
    edge_exp<<<BE, 256, 0, stream>>>(ei, logits, mkey, denom);
    edge_scatter<<<BS, 256, 0, stream>>>(ei, logits, denom, agg, out);
    bias_add<<<BN, 256, 0, stream>>>(out, b2);

    (void)in_sizes; (void)n_in; (void)ws_size;
}
